// VectorQuantizer_48009144435371
// MI455X (gfx1250) — matrix operand order repaired and checked by reading, not compiled, not run on hardware
//
#include <hip/hip_runtime.h>

typedef __attribute__((ext_vector_type(16))) __bf16 v16bf;
typedef __attribute__((ext_vector_type(8)))  float  v8f;
typedef __attribute__((ext_vector_type(4)))  float  v4f;

#define E_DIM   256
#define N_EMBED 1024
#define N_ROWS  (32*16*16*16)      /* 131072 rows of z_flat */
#define Z_TOTAL (N_ROWS*E_DIM)     /* 33554432 elements of z / z_q */

// ---------------------------------------------------------------------------
// Kernel 1: convert emb (fp32) -> bf16 in workspace (512 KB, stays hot in L2)
// ---------------------------------------------------------------------------
__global__ void vq_cvt_emb(const float* __restrict__ emb, __bf16* __restrict__ out) {
    int i = blockIdx.x * 256 + threadIdx.x;   // grid sized exactly
    out[i] = (__bf16)emb[i];
}

// ---------------------------------------------------------------------------
// Helper: load one 16-row x K=32 A-chunk of z (fp32) as bf16 in WMMA A layout.
//   lanes 0-15 (half=0): row M=l16, elems 0..15 <-> k0+0..15
//   lanes 16-31(half=1): row M=l16, elems 0..15 <-> k0+16..31
// ---------------------------------------------------------------------------
__device__ __forceinline__ v16bf load_a_chunk(const float* zrow, int k0, int half) {
    const v4f* p0 = (const v4f*)(zrow + k0 + half * 16);
    const v4f* p1 = (const v4f*)(zrow + k0 + half * 16 + 8);
    v4f f0 = p0[0], f1 = p0[1];
    v4f f2 = p1[0], f3 = p1[1];
    v16bf a;
#pragma unroll
    for (int e = 0; e < 4; ++e) {
        a[e]      = (__bf16)f0[e];
        a[4 + e]  = (__bf16)f1[e];
        a[8 + e]  = (__bf16)f2[e];
        a[12 + e] = (__bf16)f3[e];
    }
    return a;
}

// ---------------------------------------------------------------------------
// Kernel 2: fused bf16 WMMA GEMM (z @ emb^T) + in-register row argmax.
// 8 waves / block; each wave owns TWO 16-row M-tiles (32 rows) so every
// loaded B tile feeds two WMMAs -> halves codebook (B) bandwidth, 32 FLOP/B.
// The 512 MB similarity matrix is never materialized.
// ---------------------------------------------------------------------------
__global__ __launch_bounds__(256) void vq_argmax_wmma(
    const float* __restrict__ z, const __bf16* __restrict__ embb,
    int* __restrict__ idx_out, float* __restrict__ idxf_out) {

    const int lane = threadIdx.x & 31;
    const int wave = threadIdx.x >> 5;
    const int half = lane >> 4;       // 0: lanes 0-15, 1: lanes 16-31
    const int l16  = lane & 15;
    const int rowBase = blockIdx.x * 256 + wave * 32;

    // ---- Load both A tiles (rows rowBase..+15 and rowBase+16..+31), K=256.
    const float* zrow0 = z + (size_t)(rowBase + l16) * E_DIM;
    const float* zrow1 = z + (size_t)(rowBase + 16 + l16) * E_DIM;
    v16bf a0[8], a1[8];
#pragma unroll
    for (int c = 0; c < 8; ++c) {
        a0[c] = load_a_chunk(zrow0, c * 32, half);
        a1[c] = load_a_chunk(zrow1, c * 32, half);
    }

    // Running per-lane argmax state. D layout: VGPR r, lanes 0-15 -> row r,
    // lanes 16-31 -> row r+8; column N = l16 within the current N-tile.
    float bestVal0[8], bestVal1[8];
    int   bestIdx0[8], bestIdx1[8];
#pragma unroll
    for (int r = 0; r < 8; ++r) {
        bestVal0[r] = -3.0e38f; bestIdx0[r] = 0;
        bestVal1[r] = -3.0e38f; bestIdx1[r] = 0;
    }

    // ---- Sweep all 64 codebook N-tiles.
    for (int nt = 0; nt < N_EMBED / 16; ++nt) {
        // B layout (16-bit, 32x16 tile, K-chunk c): lane l16 = column N,
        // half=0 -> k = c*32 + 0..15, half=1 -> k = c*32 + 16..31 (contiguous).
        const __bf16* brow = embb + (size_t)(nt * 16 + l16) * E_DIM + half * 16;

        v8f acc0 = {}, acc1 = {};
        // Two batched passes of 4 chunks: loads clause together, waits spread.
#pragma unroll
        for (int p = 0; p < 2; ++p) {
            v16bf b[4];
#pragma unroll
            for (int c = 0; c < 4; ++c)
                b[c] = *(const v16bf*)(brow + (p * 4 + c) * 32);
#pragma unroll
            for (int c = 0; c < 4; ++c) {
                acc0 = __builtin_amdgcn_wmma_f32_16x16x32_bf16(
                    false, a0[p * 4 + c], false, b[c], (short)0, acc0, false, false);
                acc1 = __builtin_amdgcn_wmma_f32_16x16x32_bf16(
                    false, a1[p * 4 + c], false, b[c], (short)0, acc1, false, false);
            }
        }

        const int nidx = nt * 16 + l16;
#pragma unroll
        for (int r = 0; r < 8; ++r) {
            float v0 = acc0[r];
            float v1 = acc1[r];
            if (v0 > bestVal0[r]) { bestVal0[r] = v0; bestIdx0[r] = nidx; }
            if (v1 > bestVal1[r]) { bestVal1[r] = v1; bestIdx1[r] = nidx; }
        }
    }

    // ---- Cross-lane argmax reduction within each 16-lane half.
    // xor masks 1,2,4,8 never cross the 16-lane boundary in wave32.
#pragma unroll
    for (int r = 0; r < 8; ++r) {
        float v = bestVal0[r]; int id = bestIdx0[r];
#pragma unroll
        for (int m = 8; m >= 1; m >>= 1) {
            float ov = __shfl_xor(v, m, 32);
            int   oi = __shfl_xor(id, m, 32);
            if (ov > v || (ov == v && oi < id)) { v = ov; id = oi; }
        }
        bestVal0[r] = v; bestIdx0[r] = id;

        v = bestVal1[r]; id = bestIdx1[r];
#pragma unroll
        for (int m = 8; m >= 1; m >>= 1) {
            float ov = __shfl_xor(v, m, 32);
            int   oi = __shfl_xor(id, m, 32);
            if (ov > v || (ov == v && oi < id)) { v = ov; id = oi; }
        }
        bestVal1[r] = v; bestIdx1[r] = id;
    }

    // lanes 0 and 16 each hold results for 8 rows of each tile.
    if (l16 == 0) {
#pragma unroll
        for (int r = 0; r < 8; ++r) {
            int row0 = rowBase + half * 8 + r;
            int row1 = rowBase + 16 + half * 8 + r;
            idx_out[row0]  = bestIdx0[r];
            idxf_out[row0] = (float)bestIdx0[r];
            idx_out[row1]  = bestIdx1[r];
            idxf_out[row1] = (float)bestIdx1[r];
        }
    }
}

// ---------------------------------------------------------------------------
// Kernel 3: gather z_q = emb[idx] (fp32), write it, accumulate sum((z_q-z)^2)
// One block per row; 256 threads = E_DIM.
// ---------------------------------------------------------------------------
__global__ __launch_bounds__(256) void vq_gather_loss(
    const float* __restrict__ z, const float* __restrict__ emb,
    const int* __restrict__ idx, float* __restrict__ zq_out,
    float* __restrict__ sum) {

    const int row = blockIdx.x;
    const int t   = threadIdx.x;
    const int code = idx[row];

    float q  = emb[(size_t)code * E_DIM + t];
    float zv = z[(size_t)row * E_DIM + t];
    zq_out[(size_t)row * E_DIM + t] = q;

    float d = q - zv;
    float s = d * d;
#pragma unroll
    for (int m = 16; m >= 1; m >>= 1) s += __shfl_xor(s, m, 32);

    __shared__ float wsum[8];
    if ((t & 31) == 0) wsum[t >> 5] = s;
    __syncthreads();
    if (t == 0) {
        float tot = 0.f;
#pragma unroll
        for (int w = 0; w < 8; ++w) tot += wsum[w];
        atomicAdd(sum, tot);
    }
}

// ---------------------------------------------------------------------------
// Init / finalize (tiny)
// ---------------------------------------------------------------------------
__global__ void vq_init(float* sum) { if (threadIdx.x == 0) *sum = 0.f; }

__global__ void vq_final(const float* __restrict__ sum, float* __restrict__ scalars) {
    if (threadIdx.x == 0) {
        float mean = *sum * (1.0f / (float)Z_TOTAL);
        scalars[0] = mean;          // vq_loss
        scalars[1] = 0.25f * mean;  // commitment_loss (BETA = 0.25)
    }
}

// ---------------------------------------------------------------------------
extern "C" void kernel_launch(void* const* d_in, const int* in_sizes, int n_in,
                              void* d_out, int out_size, void* d_ws, size_t ws_size,
                              hipStream_t stream) {
    const float* z   = (const float*)d_in[0];   // [131072, 256] fp32
    const float* emb = (const float*)d_in[1];   // [1024, 256] fp32

    float* out     = (float*)d_out;
    float* zq      = out;                        // 33554432 floats
    float* scalars = out + Z_TOTAL;              // vq_loss, commitment_loss
    float* idxf    = out + Z_TOTAL + 2;          // 131072 indices as float

    char*   ws   = (char*)d_ws;
    __bf16* embb = (__bf16*)ws;                                           // 512 KB
    int*    idx  = (int*)(ws + (size_t)N_EMBED * E_DIM * sizeof(__bf16)); // 512 KB
    float*  sum  = (float*)(ws + (size_t)N_EMBED * E_DIM * sizeof(__bf16)
                               + (size_t)N_ROWS * sizeof(int));

    vq_init<<<1, 32, 0, stream>>>(sum);
    vq_cvt_emb<<<(N_EMBED * E_DIM) / 256, 256, 0, stream>>>(emb, embb);
    vq_argmax_wmma<<<N_ROWS / 256, 256, 0, stream>>>(z, embb, idx, idxf);
    vq_gather_loss<<<N_ROWS, 256, 0, stream>>>(z, emb, idx, zq, sum);
    vq_final<<<1, 32, 0, stream>>>(sum, scalars);
}
